// _Attention_37288906064242
// MI455X (gfx1250) — matrix operand order repaired and checked by reading, not compiled, not run on hardware
//
#include <hip/hip_runtime.h>
#include <hip/hip_bf16.h>

typedef __attribute__((ext_vector_type(16))) _Float16 v16h;
typedef __attribute__((ext_vector_type(8)))  _Float16 v8h;
typedef __attribute__((ext_vector_type(8)))  float    v8f;

#define DIM      768
#define NQKV     2304
#define SEQ      4096
#define BATCH    2
#define HEADS    12
#define HD       64
#define NTOK     (BATCH * SEQ)   // 8192

// ---------------------------------------------------------------------------
// Tensor Data Mover support (this toolchain: 6-arg builtin, clang-23 form)
// ---------------------------------------------------------------------------
#if defined(__AMDGCN__) && __has_builtin(__builtin_amdgcn_tensor_load_to_lds) && \
    __has_builtin(__builtin_amdgcn_s_wait_tensorcnt)
#define HAVE_TDM 1
#else
#define HAVE_TDM 0
#endif

__device__ __forceinline__ unsigned lds_offset(const void* p) {
#if defined(__AMDGCN__)
    // generic -> LDS addrspace cast; AS(3) pointers are 32-bit LDS byte offsets
    return (unsigned)(unsigned long long)(const __attribute__((address_space(3))) void*)p;
#else
    return 0u;
#endif
}

#if HAVE_TDM
typedef __attribute__((ext_vector_type(4))) unsigned int v4u;
typedef __attribute__((ext_vector_type(8))) int          v8i_;
typedef __attribute__((ext_vector_type(4))) int          v4i_;

// 2D tile DMA: f16 elements, tile_w elems per row (contiguous), tile_h rows,
// row_stride elems between global rows. Destination LDS is tightly packed.
__device__ __forceinline__ void tdm_load_2d_f16(unsigned lds_off,
                                                const _Float16* gptr,
                                                int tile_w, int tile_h, int row_stride) {
    unsigned long long ga = (unsigned long long)gptr;
    v4u g0;
    g0[0] = 1u;                                              // count=1, no gather
    g0[1] = lds_off;                                         // lds_addr (bytes)
    g0[2] = (unsigned)(ga & 0xffffffffu);                    // global_addr lo
    g0[3] = (unsigned)((ga >> 32) & 0x01ffffffu) | (2u << 30); // addr hi | type=2
    v8i_ g1;
    g1[0] = (1 << 16);                                       // data_size = 2 bytes
    g1[1] = (tile_w << 16);                                  // tensor_dim0[15:0] @ [31:16]
    g1[2] = (tile_h << 16);                                  // dim0 hi(0) | tensor_dim1[15:0]
    g1[3] = (tile_w << 16);                                  // dim1 hi(0) | tile_dim0
    g1[4] = tile_h;                                          // tile_dim1 | tile_dim2=0
    g1[5] = row_stride;                                      // tensor_dim0_stride lo
    g1[6] = 0;
    g1[7] = 0;
    v4i_ z4 = {0, 0, 0, 0};
    v8i_ z8 = {0, 0, 0, 0, 0, 0, 0, 0};
    __builtin_amdgcn_tensor_load_to_lds(g0, g1, z4, z4, z8, 0);
}
#endif

// ---------------------------------------------------------------------------
// WMMA fragment loaders (CDNA5 ISA 7.12.2 layouts, wave32), 16B vector loads
// ---------------------------------------------------------------------------

// A matrix 16x32 (MxK), row-major source, leading dim ld (halves).
// lanes 0-15: M=lane, K=0..7 / 16..23 ; lanes 16-31: M=lane-16, K=8..15 / 24..31
__device__ __forceinline__ v16h load_frag_a(const _Float16* __restrict__ base, int ld) {
    const int l  = threadIdx.x & 31;
    const int m  = l & 15;
    const int kb = (l >> 4) << 3;           // 0 or 8
    const _Float16* p = base + m * ld + kb;
    v8h lo = *(const v8h*)p;
    v8h hi = *(const v8h*)(p + 16);
    v16h a;
#pragma unroll
    for (int j = 0; j < 8; ++j) { a[j] = lo[j]; a[j + 8] = hi[j]; }
    return a;
}

// B matrix 32x16 (KxN) from an NxK row-major source (B[k][n] = src[n*ld + k]):
// lanes 0-15: N=lane, K=0..7 / 16..23 ; lanes 16-31: N=lane-16, K=8..15 / 24..31
__device__ __forceinline__ v16h load_frag_b_nk(const _Float16* __restrict__ base, int ld) {
    const int l  = threadIdx.x & 31;
    const int n  = l & 15;
    const int kb = (l >> 4) << 3;           // 0 or 8
    const _Float16* p = base + n * ld + kb;
    v8h lo = *(const v8h*)p;
    v8h hi = *(const v8h*)(p + 16);
    v16h b;
#pragma unroll
    for (int j = 0; j < 8; ++j) { b[j] = lo[j]; b[j + 8] = hi[j]; }
    return b;
}

#define WMMA_F32_F16(A, B, C) \
    __builtin_amdgcn_wmma_f32_16x16x32_f16(false, (A), false, (B), (short)0, (C), false, false)

// ---------------------------------------------------------------------------
// fp32 -> f16 convert
// ---------------------------------------------------------------------------
__global__ void cvt_f32_f16(const float* __restrict__ src, _Float16* __restrict__ dst, int n) {
    int i = blockIdx.x * blockDim.x + threadIdx.x;
    if (i < n) dst[i] = (_Float16)src[i];
}

// ---------------------------------------------------------------------------
// Kernel 1: QKV GEMM  [8192,768] x [768,2304]^T + bias -> Q,K,V [b,h,n,d] f16
// Double-buffered 64x32 weight tile in LDS; TDM DMA for tile k+1 overlaps the
// 4 WMMAs of tile k. Q pre-scaled by 1/sqrt(d)=0.125.
// ---------------------------------------------------------------------------
__global__ void __launch_bounds__(256)
qkv_wmma(const _Float16* __restrict__ xh, const _Float16* __restrict__ wh,
         const float* __restrict__ bias,
         _Float16* __restrict__ Qh, _Float16* __restrict__ Kh, _Float16* __restrict__ Vh) {
    __shared__ _Float16 Bl[2][64 * 32];
    const int w = threadIdx.x >> 5;
    const int l = threadIdx.x & 31;
    const int row0 = blockIdx.x * 128 + w * 16;
    const int col0 = blockIdx.y * 64;
#if HAVE_TDM
    const unsigned bl_off[2] = { lds_offset(Bl[0]), lds_offset(Bl[1]) };
#endif
    const int rr = threadIdx.x >> 2;            // 0..63 weight row
    const int cc = (threadIdx.x & 3) * 8;       // k chunk

    // prologue: stage tile k=0 into buffer 0
#if HAVE_TDM
    if (w == 0) tdm_load_2d_f16(bl_off[0], wh + (size_t)col0 * DIM, 32, 64, DIM);
#else
    *(v8h*)(Bl[0] + rr * 32 + cc) = *(const v8h*)(wh + (size_t)(col0 + rr) * DIM + cc);
#endif

    v8f acc[4] = {};
    for (int ks = 0; ks < DIM / 32; ++ks) {
        const int cur = ks & 1, nxt = cur ^ 1;
        const int k = ks * 32;
#if HAVE_TDM
        if (w == 0) __builtin_amdgcn_s_wait_tensorcnt(0);
#endif
        __syncthreads();                        // tile(ks) ready in Bl[cur]
        if (k + 32 < DIM) {                     // start tile(ks+1) -> Bl[nxt]
#if HAVE_TDM
            if (w == 0)
                tdm_load_2d_f16(bl_off[nxt], wh + (size_t)col0 * DIM + k + 32, 32, 64, DIM);
#else
            *(v8h*)(Bl[nxt] + rr * 32 + cc) =
                *(const v8h*)(wh + (size_t)(col0 + rr) * DIM + k + 32 + cc);
#endif
            __builtin_prefetch(xh + (size_t)row0 * DIM + k + 32, 0, 0);
        }
        v16h a = load_frag_a(xh + (size_t)row0 * DIM + k, DIM);
#pragma unroll
        for (int t = 0; t < 4; ++t) {
            v16h b = load_frag_b_nk(Bl[cur] + (t * 16) * 32, 32);
            acc[t] = WMMA_F32_F16(a, b, acc[t]);
        }
    }

    const int mbase = row0 + ((l >> 4) << 3);
    const int nbase = col0 + (l & 15);
#pragma unroll
    for (int t = 0; t < 4; ++t) {
        const int f    = nbase + t * 16;
        const int tsel = f / DIM;               // 0=q 1=k 2=v
        const int rem  = f % DIM;
        const int hh   = rem >> 6;
        const int d    = rem & 63;
        _Float16* dst = (tsel == 0) ? Qh : ((tsel == 1) ? Kh : Vh);
        const float bv  = bias[f];
        const float scl = (tsel == 0) ? 0.125f : 1.0f;
#pragma unroll
        for (int r = 0; r < 8; ++r) {
            const int gm = mbase + r;
            const int bb = gm >> 12;
            const int nn = gm & 4095;
            float v = (acc[t][r] + bv) * scl;
            dst[((size_t)(bb * HEADS + hh) * SEQ + nn) * HD + d] = (_Float16)v;
        }
    }
}

// ---------------------------------------------------------------------------
// Kernel 2: flash attention per (b,h). 256 thr (8 waves), 128 q-rows/block.
// Double-buffered K (TDM DMA) and V (register-staged, transposed store) tiles;
// tile i+1 is in flight underneath the S/softmax/PV work of tile i.
// ---------------------------------------------------------------------------
__global__ void __launch_bounds__(256)
attn_wmma(const _Float16* __restrict__ Qh, const _Float16* __restrict__ Kh,
          const _Float16* __restrict__ Vh, _Float16* __restrict__ Oh) {
    __shared__ _Float16 Kl[2][64 * HD];       // [key][d]
    __shared__ _Float16 Vt[2][HD * 64];       // [d][key] (transposed)
    __shared__ _Float16 Pl[8][16 * 64];

    const int w  = threadIdx.x >> 5;
    const int l  = threadIdx.x & 31;
    const int bh = blockIdx.y;
    const int b  = bh / HEADS;
    const int h  = bh % HEADS;
    const size_t head = (size_t)bh * SEQ * HD;
    const _Float16* Qp = Qh + head;
    const _Float16* Kp = Kh + head;
    const _Float16* Vp = Vh + head;
#if HAVE_TDM
    const unsigned kl_off[2] = { lds_offset(Kl[0]), lds_offset(Kl[1]) };
#endif

    const int q0 = blockIdx.x * 128 + w * 16;
    v16h aq0 = load_frag_a(Qp + (size_t)q0 * HD + 0,  HD);
    v16h aq1 = load_frag_a(Qp + (size_t)q0 * HD + 32, HD);

    v8f o[4] = {};
    float mi[8], li[8];
#pragma unroll
    for (int r = 0; r < 8; ++r) { mi[r] = -1e30f; li[r] = 0.0f; }

    const int lrow = threadIdx.x >> 2;        // 0..63 key row
    const int lcol = (threadIdx.x & 3) * 16;  // d chunk start
    _Float16* Pw = Pl[w];

    // prologue: stage tile 0
#if HAVE_TDM
    if (w == 0) tdm_load_2d_f16(kl_off[0], Kp, HD, 64, HD);
#else
    {
        const v8h* gk = (const v8h*)(Kp + (size_t)lrow * HD + lcol);
        v8h* sk = (v8h*)(Kl[0] + lrow * HD + lcol);
        sk[0] = gk[0]; sk[1] = gk[1];
    }
#endif
    {
        v8h a0 = *(const v8h*)(Vp + (size_t)lrow * HD + lcol);
        v8h a1 = *(const v8h*)(Vp + (size_t)lrow * HD + lcol + 8);
#pragma unroll
        for (int j = 0; j < 8; ++j) {
            Vt[0][(lcol + j) * 64 + lrow]     = a0[j];
            Vt[0][(lcol + 8 + j) * 64 + lrow] = a1[j];
        }
    }

    for (int it = 0; it < SEQ / 64; ++it) {
        const int cur = it & 1, nxt = cur ^ 1;
        const int k0 = it * 64;
        const bool more = (k0 + 64 < SEQ);
#if HAVE_TDM
        if (w == 0) __builtin_amdgcn_s_wait_tensorcnt(0);
#endif
        __syncthreads();                      // tile(it) ready, buffers(nxt) free

        v8h vr0 = {}, vr1 = {};
        if (more) {
#if HAVE_TDM
            if (w == 0)                       // DMA K(it+1) under compute
                tdm_load_2d_f16(kl_off[nxt], Kp + (size_t)(k0 + 64) * HD, HD, 64, HD);
#else
            const v8h* gk = (const v8h*)(Kp + (size_t)(k0 + 64 + lrow) * HD + lcol);
            v8h* sk = (v8h*)(Kl[nxt] + lrow * HD + lcol);
            sk[0] = gk[0]; sk[1] = gk[1];
#endif
            // V(it+1): global loads in flight during compute
            vr0 = *(const v8h*)(Vp + (size_t)(k0 + 64 + lrow) * HD + lcol);
            vr1 = *(const v8h*)(Vp + (size_t)(k0 + 64 + lrow) * HD + lcol + 8);
        }

        // S = Q * K^T  (16 x 64 strip; 1/sqrt(d) folded into Q)
        v8f s[4] = {};
#pragma unroll
        for (int nt = 0; nt < 4; ++nt) {
            v16h bk0 = load_frag_b_nk(Kl[cur] + (nt * 16) * HD + 0,  HD);
            s[nt] = WMMA_F32_F16(aq0, bk0, s[nt]);
            v16h bk1 = load_frag_b_nk(Kl[cur] + (nt * 16) * HD + 32, HD);
            s[nt] = WMMA_F32_F16(aq1, bk1, s[nt]);
        }

        // online softmax: C-layout row M = r + 8*(lane>=16) spans a 16-lane half
#pragma unroll
        for (int r = 0; r < 8; ++r) {
            float mx = fmaxf(fmaxf(s[0][r], s[1][r]), fmaxf(s[2][r], s[3][r]));
#pragma unroll
            for (int msk = 1; msk < 16; msk <<= 1)
                mx = fmaxf(mx, __shfl_xor(mx, msk, 32));
            const float mnew = fmaxf(mi[r], mx);
            const float corr = __expf(mi[r] - mnew);
            const int   m    = r + ((l >> 4) << 3);
            float rs = 0.0f;
#pragma unroll
            for (int nt = 0; nt < 4; ++nt) {
                float p = __expf(s[nt][r] - mnew);
                rs += p;
                Pw[m * 64 + nt * 16 + (l & 15)] = (_Float16)p;
                o[nt][r] = o[nt][r] * corr;
            }
#pragma unroll
            for (int msk = 1; msk < 16; msk <<= 1)
                rs += __shfl_xor(rs, msk, 32);
            li[r] = li[r] * corr + rs;
            mi[r] = mnew;
        }
        __syncthreads();                      // P visible; S-stage reads done

        // O += P * V   (B-frags from transposed Vt are contiguous)
#pragma unroll
        for (int dt = 0; dt < 4; ++dt) {
#pragma unroll
            for (int ks = 0; ks < 2; ++ks) {
                v16h ap = load_frag_a(Pw + ks * 32, 64);
                v16h bv = load_frag_b_nk(Vt[cur] + (dt * 16) * 64 + ks * 32, 64);
                o[dt] = WMMA_F32_F16(ap, bv, o[dt]);
            }
        }

        // stage V(it+1) into the other buffer (race-free past the P barrier)
        if (more) {
#pragma unroll
            for (int j = 0; j < 8; ++j) {
                Vt[nxt][(lcol + j) * 64 + lrow]     = vr0[j];
                Vt[nxt][(lcol + 8 + j) * 64 + lrow] = vr1[j];
            }
        }
    }

    // epilogue: O /= l, scatter to [b, n, h*64+d] f16
#pragma unroll
    for (int r = 0; r < 8; ++r) {
        const float inv  = 1.0f / li[r];
        const int   nrow = q0 + r + ((l >> 4) << 3);
        const size_t base = ((size_t)(b * SEQ + nrow)) * DIM + h * HD;
#pragma unroll
        for (int dt = 0; dt < 4; ++dt)
            Oh[base + dt * 16 + (l & 15)] = (_Float16)(o[dt][r] * inv);
    }
}

// ---------------------------------------------------------------------------
// Kernel 3: proj GEMM  [8192,768] x [768,768]^T + bias -> fp32 out
// Same double-buffered pipeline as kernel 1.
// ---------------------------------------------------------------------------
__global__ void __launch_bounds__(256)
proj_wmma(const _Float16* __restrict__ Oh, const _Float16* __restrict__ wh,
          const float* __restrict__ bias, float* __restrict__ out) {
    __shared__ _Float16 Bl[2][64 * 32];
    const int w = threadIdx.x >> 5;
    const int l = threadIdx.x & 31;
    const int row0 = blockIdx.x * 128 + w * 16;
    const int col0 = blockIdx.y * 64;
#if HAVE_TDM
    const unsigned bl_off[2] = { lds_offset(Bl[0]), lds_offset(Bl[1]) };
#endif
    const int rr = threadIdx.x >> 2;
    const int cc = (threadIdx.x & 3) * 8;

#if HAVE_TDM
    if (w == 0) tdm_load_2d_f16(bl_off[0], wh + (size_t)col0 * DIM, 32, 64, DIM);
#else
    *(v8h*)(Bl[0] + rr * 32 + cc) = *(const v8h*)(wh + (size_t)(col0 + rr) * DIM + cc);
#endif

    v8f acc[4] = {};
    for (int ks = 0; ks < DIM / 32; ++ks) {
        const int cur = ks & 1, nxt = cur ^ 1;
        const int k = ks * 32;
#if HAVE_TDM
        if (w == 0) __builtin_amdgcn_s_wait_tensorcnt(0);
#endif
        __syncthreads();
        if (k + 32 < DIM) {
#if HAVE_TDM
            if (w == 0)
                tdm_load_2d_f16(bl_off[nxt], wh + (size_t)col0 * DIM + k + 32, 32, 64, DIM);
#else
            *(v8h*)(Bl[nxt] + rr * 32 + cc) =
                *(const v8h*)(wh + (size_t)(col0 + rr) * DIM + k + 32 + cc);
#endif
            __builtin_prefetch(Oh + (size_t)row0 * DIM + k + 32, 0, 0);
        }
        v16h a = load_frag_a(Oh + (size_t)row0 * DIM + k, DIM);
#pragma unroll
        for (int t = 0; t < 4; ++t) {
            v16h b = load_frag_b_nk(Bl[cur] + (t * 16) * 32, 32);
            acc[t] = WMMA_F32_F16(a, b, acc[t]);
        }
    }

    const int mbase = row0 + ((l >> 4) << 3);
    const int nbase = col0 + (l & 15);
#pragma unroll
    for (int t = 0; t < 4; ++t) {
        const int f  = nbase + t * 16;
        const float bv = bias[f];
#pragma unroll
        for (int r = 0; r < 8; ++r)
            out[(size_t)(mbase + r) * DIM + f] = acc[t][r] + bv;
    }
}

// ---------------------------------------------------------------------------
// Launcher
// ---------------------------------------------------------------------------
extern "C" void kernel_launch(void* const* d_in, const int* in_sizes, int n_in,
                              void* d_out, int out_size, void* d_ws, size_t ws_size,
                              hipStream_t stream) {
    const float* x      = (const float*)d_in[0];
    const float* qkv_w  = (const float*)d_in[1];
    const float* qkv_b  = (const float*)d_in[2];
    const float* proj_w = (const float*)d_in[3];
    const float* proj_b = (const float*)d_in[4];
    float* out = (float*)d_out;

    _Float16* xh     = (_Float16*)d_ws;
    _Float16* wqkvh  = xh     + (size_t)NTOK * DIM;
    _Float16* wprojh = wqkvh  + (size_t)NQKV * DIM;
    _Float16* Qh     = wprojh + (size_t)DIM * DIM;
    _Float16* Kh     = Qh     + (size_t)BATCH * HEADS * SEQ * HD;
    _Float16* Vh     = Kh     + (size_t)BATCH * HEADS * SEQ * HD;
    _Float16* Oh     = Vh     + (size_t)BATCH * HEADS * SEQ * HD;

    const int nx = NTOK * DIM, nwq = NQKV * DIM, nwp = DIM * DIM;
    cvt_f32_f16<<<(nx  + 255) / 256, 256, 0, stream>>>(x,      xh,     nx);
    cvt_f32_f16<<<(nwq + 255) / 256, 256, 0, stream>>>(qkv_w,  wqkvh,  nwq);
    cvt_f32_f16<<<(nwp + 255) / 256, 256, 0, stream>>>(proj_w, wprojh, nwp);

    qkv_wmma <<<dim3(NTOK / 128, NQKV / 64), 256, 0, stream>>>(xh, wqkvh, qkv_b, Qh, Kh, Vh);
    attn_wmma<<<dim3(SEQ / 128, BATCH * HEADS), 256, 0, stream>>>(Qh, Kh, Vh, Oh);
    proj_wmma<<<dim3(NTOK / 128, DIM / 64), 256, 0, stream>>>(Oh, wprojh, proj_b, out);
}
